// net_807453851732
// MI455X (gfx1250) — matrix operand order repaired and checked by reading, not compiled, not run on hardware
//
#include <hip/hip_runtime.h>
#include <math.h>

// Problem constants (from the reference)
#define NN 8192
#define DD 128
#define EMBD 64
#define IN_DIM 896

typedef __attribute__((ext_vector_type(16))) __bf16 v16bf;
typedef __attribute__((ext_vector_type(8)))  __bf16 v8bf;
typedef __attribute__((ext_vector_type(8)))  float  v8f;
typedef __attribute__((ext_vector_type(4)))  float  v4f;

// ---------------------------------------------------------------------------
// CDNA5 async global->LDS copy (ASYNCcnt-tracked, bypasses VGPRs)
// ---------------------------------------------------------------------------
__device__ inline void async_copy_b128(unsigned lds_off, const void* gaddr) {
    asm volatile("global_load_async_to_lds_b128 %0, %1, off"
                 :: "v"(lds_off), "v"(gaddr) : "memory");
}
__device__ inline void wait_async0() {
    asm volatile("s_wait_asynccnt 0x0" ::: "memory");
}

// ---------------------------------------------------------------------------
// Small helper kernels
// ---------------------------------------------------------------------------
__global__ void zero_mult_kernel(float* mult) {
    int i = blockIdx.x * blockDim.x + threadIdx.x;
    if (i < NN) mult[i] = 0.0f;
}

__global__ void hist_kernel(const int* __restrict__ n_id, float* __restrict__ mult) {
    int i = blockIdx.x * blockDim.x + threadIdx.x;
    if (i < NN) atomicAdd(&mult[n_id[i]], 1.0f);  // exact integer counts in f32
}

// x [N][128] f32  ->  xbT [128][N] bf16 (transposed panel for GEMM B operand)
__global__ void xT_bf16_kernel(const float* __restrict__ x, __bf16* __restrict__ xbT) {
    int i = blockIdx.x * blockDim.x + threadIdx.x;   // over N*D
    int m = i >> 7, d = i & 127;
    xbT[(size_t)d * NN + m] = (__bf16)x[i];
}

__device__ inline v8bf cvt8(v4f a, v4f b) {
    v8bf v;
#pragma unroll
    for (int i = 0; i < 4; ++i) { v[i] = (__bf16)a[i]; v[4 + i] = (__bf16)b[i]; }
    return v;
}

// ---------------------------------------------------------------------------
// Main tiled WMMA GEMM:  out[m][n] = sum_k A[m][k] * BT[n][k]
//   A: f32 row-major [M x 8192], nontemporal-streamed, converted to bf16 in regs
//   BT: bf16 row-major [Ncols x 8192], staged by async global->LDS (no VGPRs)
// Double-buffered LDS, one barrier per K-stage.
// MODE 0 (diffusion): out = x - tau*acc -> xd f32 ;  vbT[n][m] = bf16(mult[m]*out)
// MODE 1 (hop1, z=k): out = acc -> u1_k f32       ;  vbT row block k*128
// MODE 2 (hop2, z=k, y=n-tile): out = acc -> u2_k f32 (256 cols)
// ---------------------------------------------------------------------------
template <int MODE>
__global__ __launch_bounds__(256)
void gemm_bf16_kernel(const float* __restrict__ Abase,
                      const __bf16* __restrict__ BT,
                      const float* __restrict__ x,
                      const float* __restrict__ tau,
                      const float* __restrict__ mult,
                      float* __restrict__ outF,
                      __bf16* __restrict__ vbT)
{
    constexpr int BK  = 64;
    constexpr int LDA = BK + 8;                 // padded LDS row (halves): conflict-free
    constexpr int OutC = (MODE == 2) ? 256 : 128;

    __shared__ __bf16 As[2][64 * LDA];          // 18.4 KB (double-buffered)
    __shared__ __bf16 Bs[2][128 * LDA];         // 36.9 KB (double-buffered)

    const int tid   = threadIdx.x;
    const int lane  = tid & 31;
    const int wid   = tid >> 5;
    const int waveM = wid >> 2;                 // 0..1
    const int waveN = wid & 3;                  // 0..3
    const int mBase = blockIdx.x * 64;
    const int nBase = (MODE == 2) ? blockIdx.y * 128 : 0;
    const int z     = blockIdx.z;

    const float* A = Abase;
    float* out = outF;
    __bf16* vb = vbT;
    if constexpr (MODE == 1) {
        A   += (size_t)z * NN * NN;
        out += (size_t)z * NN * 128;
        vb  += (size_t)z * 128 * NN;
    }
    if constexpr (MODE == 2) {
        A   += (size_t)z * NN * NN;
        out += (size_t)z * NN * 256;
    }

    // --- staging assignment ---
    // A tile 64x64 f32: 16 f32/thread through regs (needs bf16 convert)
    const int aRow = tid >> 2, aQuad = tid & 3;
    const float* gA = A + (size_t)(mBase + aRow) * NN + aQuad * 16;
    // B tile 128x64 bf16: 64 B/thread via 4 async b128 copies, no VGPR staging
    const int bRow = tid >> 1;
    const int bCol = (tid & 1) * 32;            // halves
    const __bf16* gB = BT + (size_t)(nBase + bRow) * NN + bCol;
    unsigned ldsB[2];
    ldsB[0] = (unsigned)(uintptr_t)&Bs[0][bRow * LDA + bCol];
    ldsB[1] = (unsigned)(uintptr_t)&Bs[1][bRow * LDA + bCol];

    v4f ra[4];
    v8f acc[2][2] = {};

    // ---- prologue: stage 0 ----
#pragma unroll
    for (int j = 0; j < 4; ++j) async_copy_b128(ldsB[0] + j * 16, gB + j * 8);
    {
        const v4f* pA = (const v4f*)gA;
#pragma unroll
        for (int q = 0; q < 4; ++q) ra[q] = __builtin_nontemporal_load(pA + q);
    }
    *(v8bf*)&As[0][aRow * LDA + aQuad * 16]     = cvt8(ra[0], ra[1]);
    *(v8bf*)&As[0][aRow * LDA + aQuad * 16 + 8] = cvt8(ra[2], ra[3]);
    wait_async0();
    __syncthreads();

    const int KT = NN / BK;                     // 128 stages
    for (int kt = 0; kt < KT; ++kt) {
        const int cur = kt & 1, nxt = cur ^ 1;
        const bool more = (kt + 1 < KT);

        // ---- issue next-stage loads (overlap with WMMA below) ----
        if (more) {
#pragma unroll
            for (int j = 0; j < 4; ++j)
                async_copy_b128(ldsB[nxt] + j * 16, gB + (size_t)(kt + 1) * BK + j * 8);
            const v4f* pA = (const v4f*)(gA + (size_t)(kt + 1) * BK);
#pragma unroll
            for (int q = 0; q < 4; ++q) ra[q] = __builtin_nontemporal_load(pA + q);
        }

        // ---- compute: 2 k-steps of 32, 2x2 WMMA tiles per wave ----
#pragma unroll
        for (int ks = 0; ks < 2; ++ks) {
            const int kA = ks * 32 + (lane >> 4) * 8;   // A lane chunk base (ISA layout)
            const int kB = ks * 32 + (lane >> 4) * 8;   // B lane chunk base
            v16bf aF[2], bF[2];
#pragma unroll
            for (int tm = 0; tm < 2; ++tm) {
                const int m0 = waveM * 32 + tm * 16 + (lane & 15);
                v8bf lo = *(const v8bf*)&As[cur][m0 * LDA + kA];
                v8bf hi = *(const v8bf*)&As[cur][m0 * LDA + kA + 16];
                aF[tm] = __builtin_shufflevector(lo, hi, 0,1,2,3,4,5,6,7,8,9,10,11,12,13,14,15);
            }
#pragma unroll
            for (int tn = 0; tn < 2; ++tn) {
                const int n0 = waveN * 32 + tn * 16 + (lane & 15);
                v8bf lo = *(const v8bf*)&Bs[cur][n0 * LDA + kB];
                v8bf hi = *(const v8bf*)&Bs[cur][n0 * LDA + kB + 16];
                bF[tn] = __builtin_shufflevector(lo, hi, 0,1,2,3,4,5,6,7,8,9,10,11,12,13,14,15);
            }
#pragma unroll
            for (int tm = 0; tm < 2; ++tm)
#pragma unroll
                for (int tn = 0; tn < 2; ++tn)
                    acc[tm][tn] = __builtin_amdgcn_wmma_f32_16x16x32_bf16(
                        false, aF[tm], false, bF[tn], (short)0, acc[tm][tn], false, false);
        }

        // ---- commit next A tile to LDS (global loads have had compute to land) ----
        if (more) {
            *(v8bf*)&As[nxt][aRow * LDA + aQuad * 16]     = cvt8(ra[0], ra[1]);
            *(v8bf*)&As[nxt][aRow * LDA + aQuad * 16 + 8] = cvt8(ra[2], ra[3]);
        }
        wait_async0();
        __syncthreads();
    }

    // ---- epilogue: C layout lane->(m = r + 8*(lane>>4), n = lane&15) ----
    const int hi16  = lane >> 4;
    const int nlane = lane & 15;
    float tau0 = 0.0f;
    if constexpr (MODE == 0) tau0 = tau[0];

#pragma unroll
    for (int tm = 0; tm < 2; ++tm) {
        const int m = mBase + waveM * 32 + tm * 16 + 8 * hi16;   // 8 consecutive rows
        float mv[8];
        if constexpr (MODE != 2) {
#pragma unroll
            for (int r = 0; r < 8; ++r) mv[r] = mult[m + r];
        }
#pragma unroll
        for (int tn = 0; tn < 2; ++tn) {
            const int n = nBase + waveN * 32 + tn * 16 + nlane;
            float val[8];
#pragma unroll
            for (int r = 0; r < 8; ++r) {
                float a = acc[tm][tn][r];
                if constexpr (MODE == 0) a = x[(size_t)(m + r) * DD + n] - tau0 * a;
                out[(size_t)(m + r) * OutC + n] = a;
                val[r] = a;
            }
            if constexpr (MODE != 2) {
                // fused next-hop B operand: vbT[n][m..m+7] = bf16(mult*val), one b128 store
                v8bf p;
#pragma unroll
                for (int r = 0; r < 8; ++r) p[r] = (__bf16)(mv[r] * val[r]);
                *(v8bf*)(vb + (size_t)n * NN + m) = p;
            }
        }
    }
}

// ---------------------------------------------------------------------------
// Encoder: emb[j] = tanh(feat(n_id[j]) @ W + b); 16 gathered rows per block,
// W-stationary over rows, LDS feature tile laid out [f][row] for float4 bcast.
// ---------------------------------------------------------------------------
__global__ __launch_bounds__(64)
void encoder_kernel(const float* __restrict__ xd, const float* __restrict__ u1,
                    const float* __restrict__ u2, const float* __restrict__ W,
                    const float* __restrict__ b, const int* __restrict__ n_id,
                    float* __restrict__ emb)
{
    __shared__ float feat[IN_DIM * 16];       // 57 KB
    const int e = threadIdx.x;                // 0..63 output column
    const int jBase = blockIdx.x * 16;

    for (int rr = 0; rr < 16; ++rr) {
        const int r = n_id[jBase + rr];
        for (int f = e; f < IN_DIM; f += 64) {
            float v;
            if (f < 128) {
                v = xd[(size_t)r * 128 + f];
            } else if (f < 384) {
                int t = f - 128;   // k*128 + d
                v = u1[((size_t)(t >> 7)) * NN * 128 + (size_t)r * 128 + (t & 127)];
            } else {
                int t = f - 384;   // k*256 + (k'*128 + d)
                v = u2[((size_t)(t >> 8)) * NN * 256 + (size_t)r * 256 + (t & 255)];
            }
            feat[f * 16 + rr] = v;
        }
    }
    __syncthreads();

    float acc[16];
    const float bv = b[e];
#pragma unroll
    for (int rr = 0; rr < 16; ++rr) acc[rr] = bv;

#pragma unroll 4
    for (int f = 0; f < IN_DIM; ++f) {
        const float w = W[f * EMBD + e];                 // 256B coalesced per step
        const float4* fp = (const float4*)&feat[f * 16]; // LDS broadcast reads
#pragma unroll
        for (int q = 0; q < 4; ++q) {
            float4 c = fp[q];
            acc[q * 4 + 0] += c.x * w;
            acc[q * 4 + 1] += c.y * w;
            acc[q * 4 + 2] += c.z * w;
            acc[q * 4 + 3] += c.w * w;
        }
    }
#pragma unroll
    for (int rr = 0; rr < 16; ++rr)
        emb[(size_t)(jBase + rr) * EMBD + e] = tanhf(acc[rr]);
}

// ---------------------------------------------------------------------------
// Launch. Workspace layout (~37 MB), all offsets 256B aligned.
// ---------------------------------------------------------------------------
extern "C" void kernel_launch(void* const* d_in, const int* in_sizes, int n_in,
                              void* d_out, int out_size, void* d_ws, size_t ws_size,
                              hipStream_t stream)
{
    const float* x    = (const float*)d_in[0];  // [N,128]
    const float* K    = (const float*)d_in[1];  // [2,N,N]
    const float* L    = (const float*)d_in[2];  // [N,N]
    const float* tau  = (const float*)d_in[3];  // [1]
    const float* Wenc = (const float*)d_in[4];  // [896,64]
    const float* benc = (const float*)d_in[5];  // [64]
    const int*   n_id = (const int*)d_in[6];    // [N]
    float* emb = (float*)d_out;                 // [N,64]

    char* ws = (char*)d_ws;
    float*  mult = (float*)ws;   ws += 32768;                    // [N] counts
    __bf16* xbT  = (__bf16*)ws;  ws += (size_t)DD * NN * 2;      // [128][N]
    float*  xd   = (float*)ws;   ws += (size_t)NN * DD * 4;      // [N][128]
    __bf16* v0bT = (__bf16*)ws;  ws += (size_t)DD * NN * 2;      // [128][N]
    float*  u1   = (float*)ws;   ws += (size_t)2 * NN * DD * 4;  // [2][N][128]
    __bf16* v1bT = (__bf16*)ws;  ws += (size_t)256 * NN * 2;     // [256][N]
    float*  u2   = (float*)ws;   ws += (size_t)2 * NN * 256 * 4; // [2][N][256]
    (void)in_sizes; (void)n_in; (void)out_size; (void)ws_size;

    zero_mult_kernel<<<NN / 256, 256, 0, stream>>>(mult);
    hist_kernel<<<NN / 256, 256, 0, stream>>>(n_id, mult);
    xT_bf16_kernel<<<(NN * DD) / 256, 256, 0, stream>>>(x, xbT);

    // xd = x - tau*(L@x); v0bT = (mult*xd)^T in bf16
    gemm_bf16_kernel<0><<<dim3(NN / 64, 1, 1), 256, 0, stream>>>(L, xbT, x, tau, mult, xd, v0bT);
    // u1_k = K[k] @ v0 ; v1bT row-block k = (mult*u1_k)^T
    gemm_bf16_kernel<1><<<dim3(NN / 64, 1, 2), 256, 0, stream>>>(K, v0bT, nullptr, nullptr, mult, u1, v1bT);
    // u2_k = K[k] @ v1 (256 cols)
    gemm_bf16_kernel<2><<<dim3(NN / 64, 2, 2), 256, 0, stream>>>(K, v1bT, nullptr, nullptr, nullptr, u2, nullptr);

    encoder_kernel<<<NN / 16, 64, 0, stream>>>(xd, u1, u2, Wenc, benc, n_id, emb);
}
